// LePEAttention_46239617908836
// MI455X (gfx1250) — matrix operand order repaired and checked by reading, not compiled, not run on hardware
//
#include <hip/hip_runtime.h>
#include <hip/hip_bf16.h>
#include <stdint.h>

typedef __attribute__((ext_vector_type(16))) _Float16 v16h;
typedef __attribute__((ext_vector_type(8)))  float    v8f;

union F16x16 { v16h v; _Float16 h[16]; };

#define HSP 32
#define WSP 8
#define NH  2
#define HD  32
#define NTOK 256
#define CCH 64
// H = W = 256, HW = 65536 (1<<16), all compile-time.

__device__ __forceinline__ v8f wmma_f16(v16h a, v16h b, v8f c) {
    return __builtin_amdgcn_wmma_f32_16x16x32_f16(false, a, false, b, (short)0, c, false, false);
}

// CDNA5 async copy: global -> LDS, 16B per lane, tracked by ASYNCcnt.
__device__ __forceinline__ void async_ld_b128(uint32_t lds_off, const float* gaddr) {
    asm volatile("global_load_async_to_lds_b128 %0, %1, off"
                 :: "v"(lds_off), "v"(gaddr) : "memory");
}
__device__ __forceinline__ void wait_async0() {
    asm volatile("s_wait_asynccnt 0x0" ::: "memory");
}

// ---------------------------------------------------------------------------
// Kernel 1: fused 1x1 convs  (q|k|v) = W_all(192x64) @ x(64 x B*H*W) + bias
// WMMA GEMM: A = weights (M=192,K=64), B = x pixels (K=64,N=16/tile)
// ---------------------------------------------------------------------------
__global__ __launch_bounds__(256) void qkv_conv_kernel(
    const float* __restrict__ x,
    const float* __restrict__ qk_w, const float* __restrict__ qk_b,
    const float* __restrict__ v_w,  const float* __restrict__ v_b,
    float* __restrict__ qb, float* __restrict__ kb, float* __restrict__ vb)
{
    __shared__ _Float16 wl[192][CCH + 2];
    __shared__ float    bias[192];

    const int tid = threadIdx.x;
    for (int i = tid; i < 192 * CCH; i += 256) {
        int o = i >> 6, c = i & 63;
        float wv = (o < 128) ? qk_w[o * CCH + c] : v_w[(o - 128) * CCH + c];
        wl[o][c] = (_Float16)wv;
    }
    for (int i = tid; i < 192; i += 256)
        bias[i] = (i < 128) ? qk_b[i] : v_b[i - 128];
    __syncthreads();

    const int lane = tid & 31;
    const int wid  = tid >> 5;
    const int mrow = lane & 15;
    const int hi   = lane >> 4;

    // 128 consecutive pixels per block; all inside one batch (HW % 128 == 0)
    const uint32_t gp0  = (uint32_t)blockIdx.x * 128u;
    const uint32_t bb   = gp0 >> 16;              // batch (scalar)
    const uint32_t base = (bb * CCH) << 16;       // (b*C)*HW
    const float* xb = x  + base;
    float* qB = qb + base;
    float* kB = kb + base;
    float* vB = vb + base;
    const uint32_t pix = (gp0 & 0xFFFFu) + (uint32_t)(wid * 16 + mrow);

    // B fragments: B[c][n] = x[b][c][pix], two K-frags (c 0..31, 32..63)
    F16x16 bx[2];
#pragma unroll
    for (int kf = 0; kf < 2; ++kf)
#pragma unroll
        for (int r = 0; r < 8; ++r)
#pragma unroll
            for (int p = 0; p < 2; ++p) {
                uint32_t c = (uint32_t)(kf * 32 + ((r & 3) * 2 + p) + ((r >> 2) * 16) + hi * 8);
                bx[kf].h[r * 2 + p] = (_Float16)xb[(c << 16) + pix];
            }

    v8f z = {};
#pragma unroll
    for (int mt = 0; mt < 12; ++mt) {
        F16x16 aw[2];
#pragma unroll
        for (int kf = 0; kf < 2; ++kf)
#pragma unroll
            for (int r = 0; r < 8; ++r)
#pragma unroll
                for (int p = 0; p < 2; ++p) {
                    int c = kf * 32 + ((r & 3) * 2 + p) + ((r >> 2) * 16) + hi * 8;
                    aw[kf].h[r * 2 + p] = wl[mt * 16 + mrow][c];
                }
        v8f d = wmma_f16(aw[0].v, bx[0].v, z);
        d     = wmma_f16(aw[1].v, bx[1].v, d);

        // D: row M = out-channel o = mt*16 + vv + 8*hi ; col N = pixel
#pragma unroll
        for (int vv = 0; vv < 8; ++vv) {
            int o = mt * 16 + vv + 8 * hi;
            float val = d[vv] + bias[o];
            float* buf; uint32_t oc;
            if (mt < 4)      { buf = qB; oc = (uint32_t)o; }
            else if (mt < 8) { buf = kB; oc = (uint32_t)(o - 64); }
            else             { buf = vB; oc = (uint32_t)(o - 128); }
            buf[(oc << 16) + pix] = val;
        }
    }
}

// ---------------------------------------------------------------------------
// Kernel 2: three 5x5 depthwise convs (zero-pad 2) + lepe fusion, 4 outputs/thread
//   lepe = dw5(v, lepe_w) + lepe_b ;  qh = dw5(q)+q5_b+lepe ;  kh = dw5(k)+k5_b+lepe
// block = 256 threads * 4 outputs = 1024 consecutive elements = 4 rows of one plane
// ---------------------------------------------------------------------------
__global__ __launch_bounds__(256) void dw5_kernel(
    const float* __restrict__ q, const float* __restrict__ k, const float* __restrict__ v,
    const float* __restrict__ q5_w, const float* __restrict__ q5_b,
    const float* __restrict__ k5_w, const float* __restrict__ k5_b,
    const float* __restrict__ lepe_w, const float* __restrict__ lepe_b,
    float* __restrict__ qh, float* __restrict__ kh, float* __restrict__ lepe)
{
    const uint32_t idx0  = (uint32_t)blockIdx.x << 10;   // *1024
    const uint32_t plane = idx0 >> 16;                   // b*64 + c (scalar)
    const int      c     = (int)(plane & 63);
    const uint32_t cbase = plane << 16;
    const int      hbase = (int)((idx0 >> 8) & 255);

    const int t  = threadIdx.x;
    const int hp = hbase + (t >> 6);
    const int w0 = (t & 63) << 2;

    const float* wq = q5_w   + c * 25;   // c uniform per block -> scalar loads
    const float* wk = k5_w   + c * 25;
    const float* wv = lepe_w + c * 25;

    float accq[4] = {}, acck[4] = {}, accl[4] = {};
#pragma unroll
    for (int dy = -2; dy <= 2; ++dy) {
        int hy = hp + dy;
        if (hy < 0 || hy > 255) continue;
        uint32_t rb = cbase + ((uint32_t)hy << 8);
        float rq[8], rk[8], rv[8];
#pragma unroll
        for (int tt = 0; tt < 8; ++tt) {
            int wx = w0 - 2 + tt;
            bool ok = (wx >= 0) && (wx < 256);
            uint32_t off = rb + (uint32_t)wx;
            rq[tt] = ok ? q[off] : 0.f;
            rk[tt] = ok ? k[off] : 0.f;
            rv[tt] = ok ? v[off] : 0.f;
        }
#pragma unroll
        for (int dx = 0; dx < 5; ++dx) {
            int wi = (dy + 2) * 5 + dx;
            float cq = wq[wi], ck = wk[wi], cv = wv[wi];
#pragma unroll
            for (int j = 0; j < 4; ++j) {
                accq[j] = fmaf(rq[j + dx], cq, accq[j]);
                acck[j] = fmaf(rk[j + dx], ck, acck[j]);
                accl[j] = fmaf(rv[j + dx], cv, accl[j]);
            }
        }
    }
    const float lb  = lepe_b[c];
    const float qb5 = q5_b[c];
    const float kb5 = k5_b[c];
    const uint32_t o = cbase + ((uint32_t)hp << 8) + (uint32_t)w0;
#pragma unroll
    for (int j = 0; j < 4; ++j) {
        float lep = accl[j] + lb;
        lepe[o + j] = lep;
        qh[o + j]   = accq[j] + qb5 + lep;
        kh[o + j]   = acck[j] + kb5 + lep;
    }
}

// ---------------------------------------------------------------------------
// Kernel 3: windowed flash attention per (window, head) pair.
//   block = 256 threads = 8 waves; wave w owns query rows w*32..w*32+31.
//   K/V tiles staged via global_load_async_to_lds_b128 (ASYNCcnt), then
//   repacked f32->f16 once; WMMA f16->f32 for QK^T and PV, online softmax.
// ---------------------------------------------------------------------------
__global__ __launch_bounds__(256) void attn_kernel(
    const float* __restrict__ qh, const float* __restrict__ kh,
    const float* __restrict__ v,  const float* __restrict__ lepe,
    float* __restrict__ out)
{
    const float SCALE = 0.17677669529663687f;   // (DIM/NUM_HEADS)^-0.5

    // W/WSP = 32 windows across, H/HSP = 8 down, 256 windows per batch
    int bid  = blockIdx.x;
    int head = bid & (NH - 1);
    int win  = bid >> 1;
    int b    = win >> 8;
    int wh   = (win >> 5) & 7;
    int ww   = win & 31;
    const int h0 = wh * HSP, w0 = ww * WSP;
    const uint32_t base_ch = ((uint32_t)(b * CCH + head * HD)) << 16;

    __shared__ float    stage[HD][NTOK + 8];     // async landing pad (f32)
    __shared__ _Float16 kt[NTOK][HD + 8];
    __shared__ _Float16 vt[NTOK][HD + 8];
    __shared__ _Float16 ps[8][32][40];

    const int tid  = threadIdx.x;
    const int lane = tid & 31;
    const int wid  = tid >> 5;
    const int mrow = lane & 15;
    const int hi   = lane >> 4;

    // ---- stage K via async DMA (4 tokens = 16B contiguous both sides) ----
    for (int i = tid; i < (NTOK * HD) / 4; i += 256) {
        int d  = i >> 6;
        int n0 = (i & 63) << 2;
        uint32_t g = base_ch + ((uint32_t)d << 16) +
                     ((uint32_t)(h0 + (n0 >> 3)) << 8) + (uint32_t)(w0 + (n0 & 7));
        async_ld_b128((uint32_t)(uintptr_t)&stage[d][n0], kh + g);
    }
    wait_async0();
    __syncthreads();
    for (int i = tid; i < NTOK * HD; i += 256) {
        int n = i & 255, d = i >> 8;
        kt[n][d] = (_Float16)stage[d][n];
    }
    __syncthreads();

    // ---- stage V via async DMA, reuse landing pad ----
    for (int i = tid; i < (NTOK * HD) / 4; i += 256) {
        int d  = i >> 6;
        int n0 = (i & 63) << 2;
        uint32_t g = base_ch + ((uint32_t)d << 16) +
                     ((uint32_t)(h0 + (n0 >> 3)) << 8) + (uint32_t)(w0 + (n0 & 7));
        async_ld_b128((uint32_t)(uintptr_t)&stage[d][n0], v + g);
    }
    wait_async0();
    __syncthreads();
    for (int i = tid; i < NTOK * HD; i += 256) {
        int n = i & 255, d = i >> 8;
        vt[n][d] = (_Float16)stage[d][n];
    }
    __syncthreads();

    // ---- Q A-fragments for this wave's two 16-row tiles (pre-scaled) ----
    F16x16 aq[2];
#pragma unroll
    for (int rt = 0; rt < 2; ++rt) {
        int n  = wid * 32 + rt * 16 + mrow;
        uint32_t rowg = base_ch + ((uint32_t)(h0 + (n >> 3)) << 8) + (uint32_t)(w0 + (n & 7));
#pragma unroll
        for (int r = 0; r < 8; ++r)
#pragma unroll
            for (int p = 0; p < 2; ++p) {
                uint32_t d = (uint32_t)(((r & 3) * 2 + p) + ((r >> 2) * 16) + hi * 8);
                float qv = qh[rowg + (d << 16)] * SCALE;
                aq[rt].h[r * 2 + p] = (_Float16)qv;
            }
    }

    v8f o_[2][2];
    float m_[2][8], l_[2][8];
#pragma unroll
    for (int rt = 0; rt < 2; ++rt) {
#pragma unroll
        for (int ct = 0; ct < 2; ++ct) o_[rt][ct] = (v8f){};
#pragma unroll
        for (int vv = 0; vv < 8; ++vv) { m_[rt][vv] = -3.0e38f; l_[rt][vv] = 0.f; }
    }
    v8f z = {};

    for (int jb = 0; jb < 8; ++jb) {
        // K^T B-fragments: B[d][key], key col = jb*32 + ct*16 + mrow
        F16x16 bk[2];
#pragma unroll
        for (int ct = 0; ct < 2; ++ct) {
            int key = jb * 32 + ct * 16 + mrow;
#pragma unroll
            for (int r = 0; r < 8; ++r)
#pragma unroll
                for (int p = 0; p < 2; ++p) {
                    int d = ((r & 3) * 2 + p) + ((r >> 2) * 16) + hi * 8;
                    bk[ct].h[r * 2 + p] = kt[key][d];
                }
        }

        v8f s[2][2];
#pragma unroll
        for (int rt = 0; rt < 2; ++rt)
#pragma unroll
            for (int ct = 0; ct < 2; ++ct)
                s[rt][ct] = wmma_f16(aq[rt].v, bk[ct].v, z);

        // online softmax per owned row (row = rt*16 + vv + 8*hi)
#pragma unroll
        for (int rt = 0; rt < 2; ++rt)
#pragma unroll
            for (int vv = 0; vv < 8; ++vv) {
                float a = fmaxf(s[rt][0][vv], s[rt][1][vv]);
                a = fmaxf(a, __shfl_xor(a, 1, 32));
                a = fmaxf(a, __shfl_xor(a, 2, 32));
                a = fmaxf(a, __shfl_xor(a, 4, 32));
                a = fmaxf(a, __shfl_xor(a, 8, 32));
                float mnew = fmaxf(m_[rt][vv], a);
                float corr = __expf(m_[rt][vv] - mnew);
                float p0 = __expf(s[rt][0][vv] - mnew);
                float p1 = __expf(s[rt][1][vv] - mnew);
                float rs = p0 + p1;
                rs += __shfl_xor(rs, 1, 32);
                rs += __shfl_xor(rs, 2, 32);
                rs += __shfl_xor(rs, 4, 32);
                rs += __shfl_xor(rs, 8, 32);
                l_[rt][vv] = l_[rt][vv] * corr + rs;
                m_[rt][vv] = mnew;
                s[rt][0][vv] = p0;
                s[rt][1][vv] = p1;
                o_[rt][0][vv] *= corr;
                o_[rt][1][vv] *= corr;
            }

        // repack P: D-layout (f32) -> A-layout (f16) via per-wave LDS scratch
#pragma unroll
        for (int rt = 0; rt < 2; ++rt)
#pragma unroll
            for (int ct = 0; ct < 2; ++ct)
#pragma unroll
                for (int vv = 0; vv < 8; ++vv)
                    ps[wid][rt * 16 + vv + 8 * hi][ct * 16 + mrow] =
                        (_Float16)s[rt][ct][vv];

        F16x16 pa[2], bv[2];
#pragma unroll
        for (int rt = 0; rt < 2; ++rt)
#pragma unroll
            for (int r = 0; r < 8; ++r)
#pragma unroll
                for (int p = 0; p < 2; ++p) {
                    int kk = ((r & 3) * 2 + p) + ((r >> 2) * 16) + hi * 8;
                    pa[rt].h[r * 2 + p] = ps[wid][rt * 16 + mrow][kk];
                }
#pragma unroll
        for (int ct = 0; ct < 2; ++ct)
#pragma unroll
            for (int r = 0; r < 8; ++r)
#pragma unroll
                for (int p = 0; p < 2; ++p) {
                    int kk = ((r & 3) * 2 + p) + ((r >> 2) * 16) + hi * 8;
                    bv[ct].h[r * 2 + p] = vt[jb * 32 + kk][ct * 16 + mrow];
                }

#pragma unroll
        for (int rt = 0; rt < 2; ++rt)
#pragma unroll
            for (int ct = 0; ct < 2; ++ct)
                o_[rt][ct] = wmma_f16(pa[rt].v, bv[ct].v, o_[rt][ct]);
    }

    // epilogue: normalize, add lepe, scatter via windows2img mapping
#pragma unroll
    for (int rt = 0; rt < 2; ++rt)
#pragma unroll
        for (int ct = 0; ct < 2; ++ct)
#pragma unroll
            for (int vv = 0; vv < 8; ++vv) {
                int n = wid * 32 + rt * 16 + vv + 8 * hi;
                uint32_t d = (uint32_t)(ct * 16 + mrow);
                uint32_t g = base_ch + (d << 16) +
                             ((uint32_t)(h0 + (n >> 3)) << 8) + (uint32_t)(w0 + (n & 7));
                out[g] = o_[rt][ct][vv] / l_[rt][vv] + lepe[g];
            }
}

// ---------------------------------------------------------------------------
extern "C" void kernel_launch(void* const* d_in, const int* in_sizes, int n_in,
                              void* d_out, int out_size, void* d_ws, size_t ws_size,
                              hipStream_t stream) {
    const float* x      = (const float*)d_in[0];
    const float* qk_w   = (const float*)d_in[1];
    const float* qk_b   = (const float*)d_in[2];
    const float* q5_w   = (const float*)d_in[3];
    const float* q5_b   = (const float*)d_in[4];
    const float* k5_w   = (const float*)d_in[5];
    const float* k5_b   = (const float*)d_in[6];
    const float* v_w    = (const float*)d_in[7];
    const float* v_b    = (const float*)d_in[8];
    const float* lepe_w = (const float*)d_in[9];
    const float* lepe_b = (const float*)d_in[10];
    float* out = (float*)d_out;

    const int  Bb = 4, H = 256, W = 256;
    const size_t HW = (size_t)H * W;
    const size_t N1 = (size_t)Bb * CCH * HW;

    float* ws   = (float*)d_ws;
    float* qbuf = ws;
    float* kbuf = ws + 1 * N1;
    float* vbuf = ws + 2 * N1;
    float* lepe = ws + 3 * N1;
    float* qhb  = ws + 4 * N1;
    float* khb  = ws + 5 * N1;

    // 1) fused 1x1 convs: 16-pixel tile per wave, 8 waves per block
    int blocks1 = (int)((Bb * HW) / 128);               // 2048
    qkv_conv_kernel<<<blocks1, 256, 0, stream>>>(
        x, qk_w, qk_b, v_w, v_b, qbuf, kbuf, vbuf);

    // 2) depthwise 5x5 x3 + lepe fusion, 4 outputs/thread
    int blocks2 = (int)(N1 / 1024);                     // 16384
    dw5_kernel<<<blocks2, 256, 0, stream>>>(
        qbuf, kbuf, vbuf, q5_w, q5_b, k5_w, k5_b, lepe_w, lepe_b,
        qhb, khb, lepe);

    // 3) windowed flash attention: one block per (window, head)
    int blocks3 = Bb * (H / HSP) * (W / WSP) * NH;      // 2048
    attn_kernel<<<blocks3, 256, 0, stream>>>(
        qhb, khb, vbuf, lepe, out);
}
